// MultiHeadLatentAttention_86380382257335
// MI455X (gfx1250) — compile-verified, matrix operand order checked by reading, not run on hardware
//
#include <hip/hip_runtime.h>
#include <hip/hip_bf16.h>

// ---- problem constants (match reference) ----
constexpr int B_    = 2;
constexpr int T_    = 2048;
constexpr int C_    = 2048;
constexpr int H_    = 16;
constexpr int NOPE_ = 128;
constexpr int ROPE_ = 64;
constexpr int VD_   = 128;
constexpr int LORA_ = 512;
constexpr int QK_   = NOPE_ + ROPE_;      // 192
constexpr int BT_   = B_ * T_;            // 4096
constexpr int KVD_  = NOPE_ + VD_;        // 256

typedef _Float16 h8   __attribute__((ext_vector_type(8)));
typedef _Float16 v16h __attribute__((ext_vector_type(16)));
typedef float    v8f  __attribute__((ext_vector_type(8)));

// Load a 16x32 WMMA A/B fragment slice from a row pointer (LDS).
// Per CDNA5 ISA: lane L holds (row/col = L%16), k-chunks {kb..kb+7, kb+16..kb+23},
// kb = (L/16)*8.  rowptr points at the start of this lane's 32-wide k-step.
__device__ __forceinline__ v16h load_frag16(const _Float16* rowptr, int kbase) {
  h8 lo = *(const h8*)(rowptr + kbase);
  h8 hi = *(const h8*)(rowptr + kbase + 16);
  v16h r;
#pragma unroll
  for (int i = 0; i < 8; ++i) { r[i] = lo[i]; r[i + 8] = hi[i]; }
  return r;
}

__device__ __forceinline__ v8f wmma16(v16h a, v16h b, v8f c) {
  return __builtin_amdgcn_wmma_f32_16x16x32_f16(false, a, false, b, (short)0, c,
                                                false, false);
}

// CDNA5 async copy: 16B global -> LDS per lane, tracked by ASYNCcnt.
// LDS byte address = low 32 bits of the generic pointer (ISA 10.2).
__device__ __forceinline__ void async_copy_b128(const void* gptr, void* lds) {
  unsigned l = (unsigned)(unsigned long long)(uintptr_t)lds;
  asm volatile("global_load_async_to_lds_b128 %0, %1, off"
               :: "v"(l), "v"(gptr)
               : "memory");
}
__device__ __forceinline__ void wait_async0() {
  asm volatile("s_wait_asynccnt 0x0" ::: "memory");
}

// ---------------- f32 -> f16 convert ----------------
__global__ void f32_to_f16(const float* __restrict__ in, _Float16* __restrict__ out,
                           long n) {
  long i = (long)blockIdx.x * blockDim.x + threadIdx.x;
  long stride = (long)gridDim.x * blockDim.x;
  for (; i < n; i += stride) out[i] = (_Float16)in[i];
}

// ---------------- tiled transpose + convert: in f32 (K x N) -> out f16 (N x K) --
__global__ void __launch_bounds__(256)
transpose_f32_f16(const float* __restrict__ in, _Float16* __restrict__ out,
                  int K, int N) {
  __shared__ float tile[32][33];
  int k0 = blockIdx.y * 32, n0 = blockIdx.x * 32;
  int tx = threadIdx.x & 31, ty = threadIdx.x >> 5;   // 32 x 8
#pragma unroll
  for (int i = 0; i < 4; ++i) {
    int k = k0 + ty + i * 8, n = n0 + tx;
    tile[ty + i * 8][tx] = (k < K && n < N) ? in[(size_t)k * N + n] : 0.f;
  }
  __syncthreads();
#pragma unroll
  for (int i = 0; i < 4; ++i) {
    int n = n0 + ty + i * 8, k = k0 + tx;
    if (n < N && k < K) out[(size_t)n * K + k] = (_Float16)tile[tx][ty + i * 8];
  }
}

// ---------------- WMMA GEMM: C[M,N] = A[M,K] * Bt[N,K]^T ----------------
// A f16 row-major [M][K], Bt f16 row-major [N][K] (pre-transposed B).
// OUT_MODE: 0 = f32 row-major, 1 = f16 row-major,
//           2 = kv split: Cv = k_nope f16 (BT x H*128 row-major),
//                         Cv2 = vT  f16 ([b][h][vd][t], 16B-vector stores)
// Block 256 threads (8 waves).  Block tile 128x128, wave tile 32x64
// (12 LDS loads per 8 WMMAs).  k-step 32, double-buffered via async-to-LDS.
template <int OUT_MODE>
__global__ void __launch_bounds__(256) __attribute__((amdgpu_waves_per_eu(4)))
gemm_f16_wmma(const _Float16* __restrict__ A, const _Float16* __restrict__ Bt,
              void* __restrict__ Cv, void* __restrict__ Cv2, int M, int N, int K) {
  __shared__ _Float16 As[2][128 * 32];   // [m][k]
  __shared__ _Float16 Bs[2][128 * 32];   // [n][k]
  const int tid  = threadIdx.x;
  const int wave = tid >> 5;
  const int wm   = wave >> 1;            // 0..3: 32-row subtile
  const int wn   = wave & 1;             // 0..1: 64-col half
  const int lane = tid & 31;
  const int half16 = lane >> 4;
  const int lcol   = lane & 15;
  const int kbase  = half16 * 8;
  const int m0 = blockIdx.y * 128;
  const int n0 = blockIdx.x * 128;
  const h8 hz = {};

  v8f acc[2][4] = {};

  auto stage_async = [&](int buf, int kt) {
#pragma unroll
    for (int i = 0; i < 2; ++i) {
      int c = tid + i * 256;
      int row = c >> 2, seg = c & 3;
      async_copy_b128(A + (size_t)(m0 + row) * K + kt + seg * 8,
                      &As[buf][row * 32 + seg * 8]);
      int n = n0 + row;
      if (n < N)
        async_copy_b128(Bt + (size_t)n * K + kt + seg * 8,
                        &Bs[buf][row * 32 + seg * 8]);
      else
        *(h8*)(&Bs[buf][row * 32 + seg * 8]) = hz;
    }
  };

  stage_async(0, 0);
  wait_async0();
  __syncthreads();

  int cur = 0;
  const int nk = K / 32;
  for (int ik = 0; ik < nk; ++ik) {
    bool has_next = (ik + 1) < nk;
    if (has_next) stage_async(cur ^ 1, (ik + 1) * 32);

    v16h a0 = load_frag16(&As[cur][(wm * 32 + lcol) * 32], kbase);
    v16h a1 = load_frag16(&As[cur][(wm * 32 + 16 + lcol) * 32], kbase);
#pragma unroll
    for (int nt = 0; nt < 4; ++nt) {
      v16h bf = load_frag16(&Bs[cur][((wn * 4 + nt) * 16 + lcol) * 32], kbase);
      acc[0][nt] = wmma16(a0, bf, acc[0][nt]);
      acc[1][nt] = wmma16(a1, bf, acc[1][nt]);
    }
    if (has_next) wait_async0();
    __syncthreads();
    cur ^= 1;
  }

  if (OUT_MODE == 2) {
#pragma unroll
    for (int mi = 0; mi < 2; ++mi) {
      const int t0 = m0 + wm * 32 + mi * 16 + 8 * half16;  // 8 consecutive t
      const int bb = t0 >> 11;                             // /T_
      const int tloc = t0 & (T_ - 1);
#pragma unroll
      for (int nt = 0; nt < 4; ++nt) {
        int col = n0 + (wn * 4 + nt) * 16 + lcol;
        int h = col >> 8, d = col & 255;
        if (d < NOPE_) {
#pragma unroll
          for (int v = 0; v < 8; ++v) {
            int row = m0 + wm * 32 + mi * 16 + v + 8 * half16;
            ((_Float16*)Cv)[(size_t)row * (H_ * NOPE_) + h * NOPE_ + d] =
                (_Float16)acc[mi][nt][v];
          }
        } else {
          h8 pk;
#pragma unroll
          for (int v = 0; v < 8; ++v) pk[v] = (_Float16)acc[mi][nt][v];
          *(h8*)((_Float16*)Cv2 +
                 (((size_t)bb * H_ + h) * VD_ + (d - NOPE_)) * T_ + tloc) = pk;
        }
      }
    }
  } else {
#pragma unroll
    for (int mi = 0; mi < 2; ++mi) {
#pragma unroll
      for (int nt = 0; nt < 4; ++nt) {
        int col = n0 + (wn * 4 + nt) * 16 + lcol;
        if (col >= N) continue;
#pragma unroll
        for (int v = 0; v < 8; ++v) {
          int row = m0 + wm * 32 + mi * 16 + v + 8 * half16;
          if (row < M) {
            if (OUT_MODE == 1)
              ((_Float16*)Cv)[(size_t)row * N + col] = (_Float16)acc[mi][nt][v];
            else
              ((float*)Cv)[(size_t)row * N + col] = acc[mi][nt][v];
          }
        }
      }
    }
  }
}

// ---------------- RoPE on q (in-place, f16) ----------------
__global__ void rope_q(_Float16* __restrict__ q, const float* __restrict__ freq,
                       int n) {
  int id = blockIdx.x * blockDim.x + threadIdx.x;
  if (id >= n) return;
  int i    = id & 31;
  int rest = id >> 5;              // (b*T+t)*H + h
  int bt   = rest >> 4;
  int t    = bt & (T_ - 1);
  size_t base = (size_t)rest * QK_ + NOPE_ + 2 * i;
  float c = freq[(size_t)(t * 32 + i) * 2 + 0];
  float s = freq[(size_t)(t * 32 + i) * 2 + 1];
  float x0 = (float)q[base], x1 = (float)q[base + 1];
  q[base]     = (_Float16)(x0 * c - x1 * s);
  q[base + 1] = (_Float16)(x1 * c + x0 * s);
}

// ---------------- RMS-norm latent + RoPE k_rope ----------------
__global__ void __launch_bounds__(128)
ckv_post(const float* __restrict__ ckv, const float* __restrict__ w,
         const float* __restrict__ freq, _Float16* __restrict__ kvlat,
         _Float16* __restrict__ kr) {
  int row = blockIdx.x;                 // b*T+t
  int t   = row & (T_ - 1);
  int tid = threadIdx.x;
  const float* src = ckv + (size_t)row * (LORA_ + ROPE_);

  float4 xv = *(const float4*)(src + tid * 4);
  float ss = xv.x * xv.x + xv.y * xv.y + xv.z * xv.z + xv.w * xv.w;
#pragma unroll
  for (int m = 1; m < 32; m <<= 1) ss += __shfl_xor(ss, m, 32);
  __shared__ float red[4];
  if ((tid & 31) == 0) red[tid >> 5] = ss;
  __syncthreads();
  float tot = red[0] + red[1] + red[2] + red[3];
  float rs = rsqrtf(tot * (1.f / (float)LORA_) + 1e-6f);

  _Float16* dst = kvlat + (size_t)row * LORA_ + tid * 4;
  dst[0] = (_Float16)(xv.x * rs * w[tid * 4 + 0]);
  dst[1] = (_Float16)(xv.y * rs * w[tid * 4 + 1]);
  dst[2] = (_Float16)(xv.z * rs * w[tid * 4 + 2]);
  dst[3] = (_Float16)(xv.w * rs * w[tid * 4 + 3]);

  if (tid < 32) {
    float c = freq[(size_t)(t * 32 + tid) * 2 + 0];
    float s = freq[(size_t)(t * 32 + tid) * 2 + 1];
    float x0 = src[LORA_ + 2 * tid], x1 = src[LORA_ + 2 * tid + 1];
    kr[(size_t)row * ROPE_ + 2 * tid]     = (_Float16)(x0 * c - x1 * s);
    kr[(size_t)row * ROPE_ + 2 * tid + 1] = (_Float16)(x1 * c + x0 * s);
  }
}

// ---------------- flash attention (causal, analytic mask) ----------------
// grid: B*H*(T/128) blocks, 256 threads.  Dynamic LDS:
//   Qs 128x192, Ks 128x192, Vs[vd][key] 128x128, Ps 8 waves x 16x64 = 144 KB
// 128-key blocks processed in two 64-key halves; Q/K/V staged via async-to-LDS.
__global__ void __launch_bounds__(256) __attribute__((amdgpu_waves_per_eu(4)))
mla_attention(const _Float16* __restrict__ q, const _Float16* __restrict__ knope,
              const _Float16* __restrict__ kr, const _Float16* __restrict__ vT,
              _Float16* __restrict__ o) {
  extern __shared__ char smem[];
  _Float16* Qs = (_Float16*)smem;          // 128*192
  _Float16* Ks = Qs + 128 * 192;           // 128*192
  _Float16* Vs = Ks + 128 * 192;           // 128*128 ([vd][key])
  _Float16* Ps = Vs + 128 * 128;           // 8*16*64

  const int nqb = T_ / 128;                // 16
  int qb = blockIdx.x % nqb;
  int bh = blockIdx.x / nqb;
  int h = bh % H_, b = bh / H_;

  const int tid = threadIdx.x, wave = tid >> 5, lane = tid & 31;
  const int half16 = lane >> 4, lcol = lane & 15, kbase = half16 * 8;
  const float scale = 0.07216878364870323f;   // 1/sqrt(192)

  // Q tile: 128 rows x 24 segs of 8 halfs (async copies; waited in first kb)
#pragma unroll
  for (int i = 0; i < 12; ++i) {
    int c = tid + i * 256;
    int row = c / 24, seg = c % 24;
    async_copy_b128(
        q + ((size_t)(b * T_ + qb * 128 + row) * H_ + h) * QK_ + seg * 8,
        Qs + row * QK_ + seg * 8);
  }

  float m_i[8], l_i[8];
  v8f o_acc[8] = {};
#pragma unroll
  for (int v = 0; v < 8; ++v) { m_i[v] = -3.0e38f; l_i[v] = 0.f; }

  const int qlast = qb * 128 + wave * 16 + 15;   // this wave's last q row

  for (int kb = 0; kb <= qb; ++kb) {
    __syncthreads();
    int k0 = kb * 128;
    // K tile: nope (row-major buffer) + broadcast rope
#pragma unroll
    for (int i = 0; i < 12; ++i) {
      int c = tid + i * 256;
      int row = c / 24, seg = c % 24, d = seg * 8;
      const _Float16* src =
          (d < NOPE_)
              ? knope + (size_t)(b * T_ + k0 + row) * (H_ * NOPE_) + h * NOPE_ + d
              : kr + (size_t)(b * T_ + k0 + row) * ROPE_ + (d - NOPE_);
      async_copy_b128(src, Ks + row * QK_ + d);
    }
    // V tile: already transposed in global ([b][h][vd][t]) -> straight copy
#pragma unroll
    for (int i = 0; i < 8; ++i) {
      int c = tid + i * 256;
      int row = c >> 4, seg = c & 15;     // row = vd, seg along keys
      async_copy_b128(vT + (((size_t)b * H_ + h) * VD_ + row) * T_ + k0 + seg * 8,
                      Vs + row * 128 + seg * 8);
    }
    wait_async0();
    __syncthreads();

    // process the 128 keys as two 64-key halves (keeps VGPR count < 256)
#pragma unroll
    for (int hh = 0; hh < 2; ++hh) {
      int kh0 = k0 + hh * 64;
      if (kh0 > qlast) continue;         // whole half causally masked for wave

      // S = Q * K^T  (16 q-rows x 64 keys), reduction dim 192
      v8f ss[4] = {};
#pragma unroll
      for (int d = 0; d < QK_; d += 32) {
        v16h a = load_frag16(Qs + (wave * 16 + lcol) * QK_ + d, kbase);
#pragma unroll
        for (int nt = 0; nt < 4; ++nt) {
          v16h bf =
              load_frag16(Ks + ((hh * 4 + nt) * 16 + lcol) * QK_ + d, kbase);
          ss[nt] = wmma16(a, bf, ss[nt]);
        }
      }

      // scale + causal mask (analytic)
#pragma unroll
      for (int nt = 0; nt < 4; ++nt) {
        int kg = kh0 + nt * 16 + lcol;
#pragma unroll
        for (int v = 0; v < 8; ++v) {
          float s = ss[nt][v] * scale;
          int qg = qb * 128 + wave * 16 + v + 8 * half16;
          if (kg > qg) s = -1.0e9f;
          ss[nt][v] = s;
        }
      }

      // online softmax per row (16-lane groups hold one row of the C layout)
#pragma unroll
      for (int v = 0; v < 8; ++v) {
        float m = -3.0e38f;
#pragma unroll
        for (int nt = 0; nt < 4; ++nt) m = fmaxf(m, ss[nt][v]);
        m = fmaxf(m, __shfl_xor(m, 1, 32));
        m = fmaxf(m, __shfl_xor(m, 2, 32));
        m = fmaxf(m, __shfl_xor(m, 4, 32));
        m = fmaxf(m, __shfl_xor(m, 8, 32));
        float m_new = fmaxf(m_i[v], m);
        float alpha = __expf(m_i[v] - m_new);
        m_i[v] = m_new;
        float lsum = 0.f;
#pragma unroll
        for (int nt = 0; nt < 4; ++nt) {
          float p = __expf(ss[nt][v] - m_new);
          lsum += p;
          Ps[(wave * 16 + v + 8 * half16) * 64 + nt * 16 + lcol] = (_Float16)p;
        }
        lsum += __shfl_xor(lsum, 1, 32);
        lsum += __shfl_xor(lsum, 2, 32);
        lsum += __shfl_xor(lsum, 4, 32);
        lsum += __shfl_xor(lsum, 8, 32);
        l_i[v] = l_i[v] * alpha + lsum;
#pragma unroll
        for (int nt = 0; nt < 8; ++nt) o_acc[nt][v] *= alpha;
      }

      // O += P * V   (key dim 64 for this half)
#pragma unroll
      for (int ks = 0; ks < 64; ks += 32) {
        v16h a = load_frag16(Ps + (wave * 16 + lcol) * 64 + ks, kbase);
#pragma unroll
        for (int nt = 0; nt < 8; ++nt) {
          v16h bf =
              load_frag16(Vs + (nt * 16 + lcol) * 128 + hh * 64 + ks, kbase);
          o_acc[nt] = wmma16(a, bf, o_acc[nt]);
        }
      }
    }
  }

  // normalize + store (b,t,h,vd)
  float inv_l[8];
#pragma unroll
  for (int v = 0; v < 8; ++v) inv_l[v] = 1.f / l_i[v];
#pragma unroll
  for (int nt = 0; nt < 8; ++nt) {
    int col = nt * 16 + lcol;
#pragma unroll
    for (int v = 0; v < 8; ++v) {
      int qg = qb * 128 + wave * 16 + v + 8 * half16;
      o[((size_t)(b * T_ + qg) * H_ + h) * VD_ + col] =
          (_Float16)(o_acc[nt][v] * inv_l[v]);
    }
  }
}

// ---------------- host ----------------
extern "C" void kernel_launch(void* const* d_in, const int* in_sizes, int n_in,
                              void* d_out, int out_size, void* d_ws, size_t ws_size,
                              hipStream_t stream) {
  (void)in_sizes; (void)n_in; (void)out_size; (void)ws_size;
  const float* x     = (const float*)d_in[0];
  const float* freq  = (const float*)d_in[1];
  const float* Wq    = (const float*)d_in[4];
  const float* Wckv  = (const float*)d_in[5];
  const float* kvw   = (const float*)d_in[6];
  const float* Wdkv  = (const float*)d_in[7];
  const float* Wproj = (const float*)d_in[8];
  float* out = (float*)d_out;

  char* ws = (char*)d_ws;
  size_t off = 0;
  auto take = [&](size_t bytes) {
    char* p = ws + off;
    off = (off + bytes + 255) & ~(size_t)255;
    return p;
  };
  _Float16* xh     = (_Float16*)take((size_t)BT_ * C_ * 2);
  _Float16* Wqt    = (_Float16*)take((size_t)C_ * H_ * QK_ * 2);          // [3072][2048]
  _Float16* Wckvt  = (_Float16*)take((size_t)C_ * (LORA_ + ROPE_) * 2);   // [576][2048]
  _Float16* Wdkvt  = (_Float16*)take((size_t)LORA_ * H_ * KVD_ * 2);      // [4096][512]
  _Float16* Wprojt = (_Float16*)take((size_t)(H_ * VD_) * C_ * 2);        // [2048][2048]
  _Float16* qh     = (_Float16*)take((size_t)BT_ * H_ * QK_ * 2);
  float*    ckvf   = (float*)take((size_t)BT_ * (LORA_ + ROPE_) * 4);
  _Float16* kvlat  = (_Float16*)take((size_t)BT_ * LORA_ * 2);
  _Float16* krh    = (_Float16*)take((size_t)BT_ * ROPE_ * 2);
  _Float16* knope  = (_Float16*)take((size_t)BT_ * H_ * NOPE_ * 2);
  _Float16* vTh    = (_Float16*)take((size_t)B_ * H_ * VD_ * T_ * 2);
  _Float16* oh     = (_Float16*)take((size_t)BT_ * H_ * VD_ * 2);

  // activations to f16; weights to f16 transposed ([N][K])
  f32_to_f16<<<2048, 256, 0, stream>>>(x, xh, (long)BT_ * C_);
  transpose_f32_f16<<<dim3((H_ * QK_ + 31) / 32, C_ / 32), 256, 0, stream>>>(
      Wq, Wqt, C_, H_ * QK_);
  transpose_f32_f16<<<dim3((LORA_ + ROPE_ + 31) / 32, C_ / 32), 256, 0, stream>>>(
      Wckv, Wckvt, C_, LORA_ + ROPE_);
  transpose_f32_f16<<<dim3((H_ * KVD_ + 31) / 32, LORA_ / 32), 256, 0, stream>>>(
      Wdkv, Wdkvt, LORA_, H_ * KVD_);
  transpose_f32_f16<<<dim3(C_ / 32, (H_ * VD_) / 32), 256, 0, stream>>>(
      Wproj, Wprojt, H_ * VD_, C_);

  // q = x @ Wq (f16 out), then RoPE in place
  gemm_f16_wmma<1><<<dim3((H_ * QK_) / 128, BT_ / 128), 256, 0, stream>>>(
      xh, Wqt, qh, nullptr, BT_, H_ * QK_, C_);
  rope_q<<<(BT_ * H_ * 32 + 255) / 256, 256, 0, stream>>>(qh, freq, BT_ * H_ * 32);

  // ckv = x @ Wckv (f32 out), then RMS-norm + k-RoPE
  gemm_f16_wmma<0><<<dim3((LORA_ + ROPE_ + 127) / 128, BT_ / 128), 256, 0, stream>>>(
      xh, Wckvt, ckvf, nullptr, BT_, LORA_ + ROPE_, C_);
  ckv_post<<<BT_, 128, 0, stream>>>(ckvf, kvw, freq, kvlat, krh);

  // kv = norm(latent) @ Wdkv, split epilogue: k_nope row-major + V transposed
  gemm_f16_wmma<2><<<dim3((H_ * KVD_) / 128, BT_ / 128), 256, 0, stream>>>(
      kvlat, Wdkvt, knope, vTh, BT_, H_ * KVD_, LORA_);

  // flash attention, 144 KB dynamic LDS per block
  size_t shmem = (size_t)(128 * 192 + 128 * 192 + 128 * 128 + 8 * 16 * 64) * 2;
  mla_attention<<<B_ * H_ * (T_ / 128), 256, shmem, stream>>>(qh, knope, krh, vTh,
                                                              oh);

  // out = O @ Wproj (f32 to d_out)
  gemm_f16_wmma<0><<<dim3(C_ / 128, BT_ / 128), 256, 0, stream>>>(
      oh, Wprojt, out, nullptr, BT_, C_, H_ * VD_);
}
